// AdvancedCausalSelfAttention_8315056685285
// MI455X (gfx1250) — matrix operand order repaired and checked by reading, not compiled, not run on hardware
//
#include <hip/hip_runtime.h>
#include <hip/hip_bf16.h>
#include <stdint.h>

#define DIM        2048
#define NUM_HEADS  16
#define NUM_KV_HEADS 4
#define HEAD_DIM   128
#define KV_DIM     512
#define BSZ        4
#define SEQ        2048
#define MROWS      (BSZ*SEQ)          // 8192
#define NQKV       (DIM + 2*KV_DIM)   // 3072

typedef __bf16 bf16_t;
typedef __attribute__((ext_vector_type(16))) __bf16 v16bf;
typedef __attribute__((ext_vector_type(8)))  float  v8f;
typedef __attribute__((ext_vector_type(4)))  unsigned int u32x4;
typedef __attribute__((ext_vector_type(4)))  unsigned int vu32x4;
typedef __attribute__((ext_vector_type(8)))  int vi32x8;
typedef __attribute__((ext_vector_type(4)))  int vi32x4;

union FragB { v16bf v; u32x4 q[2]; };

__device__ inline bf16_t f2bf(float f) {
  union { float f; unsigned int u; } in; in.f = f;
  unsigned int r = in.u + 0x7FFFu + ((in.u >> 16) & 1u);   // round-to-nearest-even
  union { unsigned short s; bf16_t b; } out; out.s = (unsigned short)(r >> 16);
  return out.b;
}
__device__ inline float bf2f(bf16_t b) {
  union { unsigned short s; bf16_t b; } in; in.b = b;
  union { unsigned int u; float f; } out; out.u = ((unsigned int)in.s) << 16;
  return out.f;
}

// ---------------------------------------------------------------- convert
__global__ __launch_bounds__(256) void cvt_f32_to_bf16(
    const float* __restrict__ src, bf16_t* __restrict__ dst, int n) {
  int i = blockIdx.x * 256 + threadIdx.x;
  if (i < n) dst[i] = f2bf(src[i]);
}

// ---------------------------------------------------------------- TDM tile load
// DMA one 128-row x 32-col bf16 tile (row stride = krow elems) into LDS,
// with HW padding of 4 DWORDs after every 16 DWORDs -> 80B LDS row stride
// (matches LD=40 bf16 elements, bank-conflict friendly).
__device__ inline void tdm_load_tile_128x32(const bf16_t* gsrc,
                                            unsigned lds_byte_off, int krow) {
  unsigned long long ga = (unsigned long long)(uintptr_t)gsrc;
  vu32x4 g0;
  g0[0] = 1u;                                              // count=1 (valid user D#)
  g0[1] = lds_byte_off;                                    // LDS destination
  g0[2] = (unsigned)ga;                                    // global_addr[31:0]
  g0[3] = ((unsigned)(ga >> 32) & 0x01FFFFFFu) | (2u << 30); // addr[56:32] | type=2
  vi32x8 g1;
  // data_size=2B (1<<16) | pad_enable (1<<20) | pad_interval=16dw (3<<22)
  // | pad_amount=4dw (3<<25)
  g1[0] = (int)((1u << 16) | (1u << 20) | (3u << 22) | (3u << 25));
  g1[1] = (int)(((unsigned)krow & 0xFFFFu) << 16);         // tensor_dim0 lo16
  g1[2] = (int)(((unsigned)krow >> 16) & 0xFFFFu);         // tensor_dim0 hi16, dim1 lo16=0
  g1[3] = (int)(16u | (32u << 16));                        // dim1 hi16 (=1<<20), tile_dim0=32
  g1[4] = 128;                                             // tile_dim1=128, tile_dim2=0
  g1[5] = krow;                                            // tensor_dim0_stride lo32
  g1[6] = 0;                                               // stride0 hi, stride1 lo
  g1[7] = 0;
  vi32x4 z4 = {0, 0, 0, 0};
#if defined(__clang_major__) && (__clang_major__ >= 23)
  vi32x8 z8 = {0, 0, 0, 0, 0, 0, 0, 0};
  __builtin_amdgcn_tensor_load_to_lds(g0, g1, z4, z4, z8, 0);
#else
  __builtin_amdgcn_tensor_load_to_lds(g0, g1, z4, z4, 0);
#endif
}

// ---------------------------------------------------------------- NT GEMM
// C[M,N] = A[M,K] * B[N,K]^T ; A,B bf16 row-major, acc f32.
// Block tile 128x128, 8 waves (2 M x 4 N), wave tile 64x32, BK=32.
// A/B tiles staged by the Tensor Data Mover (double buffered, TENSORcnt sync).
template<bool OUT_F32>
__global__ __launch_bounds__(256) void gemm_nt_bf16(
    const bf16_t* __restrict__ A, const bf16_t* __restrict__ B,
    void* __restrict__ Cv, int M, int N, int K)
{
  constexpr int LD = 40;                       // 32 + 4dw HW pad (80B rows)
  constexpr unsigned TILE_BYTES = 128u * LD * 2u;  // 10240
  __shared__ __align__(16) bf16_t As[2][128*LD];
  __shared__ __align__(16) bf16_t Bs[2][128*LD];

  const int tid  = threadIdx.x;
  const int lane = tid & 31;
  const int w    = tid >> 5;
  const int l15  = lane & 15;
  const int hi   = lane >> 4;
  const int wm   = w & 1, wn = w >> 1;
  const int m0   = blockIdx.y * 128;
  const int n0   = blockIdx.x * 128;

  const unsigned ldsA = (unsigned)(uintptr_t)&As[0][0];   // LDS byte offsets
  const unsigned ldsB = (unsigned)(uintptr_t)&Bs[0][0];

  auto stage = [&](int buf, int k0) {          // wave 0 only: 2 TDM descriptors
    tdm_load_tile_128x32(&A[(size_t)m0*K + k0], ldsA + (unsigned)buf*TILE_BYTES, K);
    tdm_load_tile_128x32(&B[(size_t)n0*K + k0], ldsB + (unsigned)buf*TILE_BYTES, K);
  };

  const v8f vz = {0.f,0.f,0.f,0.f,0.f,0.f,0.f,0.f};
  v8f acc[4][2];
#pragma unroll
  for (int mt = 0; mt < 4; ++mt)
#pragma unroll
    for (int nt = 0; nt < 2; ++nt) acc[mt][nt] = vz;

  const int nk = K >> 5;
  if (tid < 32) stage(0, 0);
  for (int kt = 0; kt < nk; ++kt) {
    const int cur = kt & 1;
    if (tid < 32) __builtin_amdgcn_s_wait_tensorcnt(0);   // cur tile DMA complete
    __syncthreads();                                      // visible to all waves
    if (kt + 1 < nk && tid < 32) stage(cur ^ 1, (kt + 1) << 5);

    FragB af[4], bfr[2];
#pragma unroll
    for (int mt = 0; mt < 4; ++mt) {           // A frag: lane=row, K chunks hi*8 / 16+hi*8
      const int base = (wm*64 + mt*16 + l15)*LD + hi*8;
      af[mt].q[0] = *(const u32x4*)&As[cur][base];
      af[mt].q[1] = *(const u32x4*)&As[cur][base + 16];
    }
#pragma unroll
    for (int nt = 0; nt < 2; ++nt) {           // B frag: lane=N col, K chunks hi*8 / 16+hi*8
      const int base = (wn*32 + nt*16 + l15)*LD + hi*8;
      bfr[nt].q[0] = *(const u32x4*)&Bs[cur][base];
      bfr[nt].q[1] = *(const u32x4*)&Bs[cur][base + 16];
    }
#pragma unroll
    for (int mt = 0; mt < 4; ++mt)
#pragma unroll
      for (int nt = 0; nt < 2; ++nt)
        acc[mt][nt] = __builtin_amdgcn_wmma_f32_16x16x32_bf16(
            false, af[mt].v, false, bfr[nt].v, (short)0, acc[mt][nt], false, false);
    __syncthreads();                           // all reads of cur done before re-stage
  }

#pragma unroll
  for (int mt = 0; mt < 4; ++mt)
#pragma unroll
    for (int nt = 0; nt < 2; ++nt)
#pragma unroll
      for (int r = 0; r < 8; ++r) {            // C layout: M = r + 8*hi, N = l15
        const int row = m0 + wm*64 + mt*16 + r + 8*hi;
        const int col = n0 + wn*32 + nt*16 + l15;
        if (OUT_F32) ((float*)Cv)[(size_t)row*N + col] = acc[mt][nt][r];
        else ((bf16_t*)Cv)[(size_t)row*N + col] = f2bf(acc[mt][nt][r]);
      }
}

// ---------------------------------------------------------------- qk-norm + rope
__device__ inline void norm_rope_row(const bf16_t* __restrict__ src,
                                     bf16_t* __restrict__ dst,
                                     int s, float gain, int lane) {
  float e0 = bf2f(src[lane]);
  float e1 = bf2f(src[lane + 32]);
  float e2 = bf2f(src[lane + 64]);
  float e3 = bf2f(src[lane + 96]);
  float ss = e0*e0 + e1*e1 + e2*e2 + e3*e3;
#pragma unroll
  for (int off = 16; off > 0; off >>= 1) ss += __shfl_xor(ss, off, 32);
  const float rms = rsqrtf(ss * (1.0f/128.0f) + 1.1920929e-07f);
  const float n0 = e0*rms, n1 = e1*rms, n2 = e2*rms, n3 = e3*rms;
  const float fs = (float)s;
  const float a0 = fs * __powf(10000.f, -(float)lane        * (1.f/64.f));
  const float a1 = fs * __powf(10000.f, -(float)(lane + 32) * (1.f/64.f));
  float c0, s0, c1, s1;
  __sincosf(a0, &s0, &c0);
  __sincosf(a1, &s1, &c1);
  dst[lane]      = f2bf(( n0*c0 + n2*s0) * gain);
  dst[lane + 64] = f2bf((-n0*s0 + n2*c0) * gain);
  dst[lane + 32] = f2bf(( n1*c1 + n3*s1) * gain);
  dst[lane + 96] = f2bf((-n1*s1 + n3*c1) * gain);
}

__global__ __launch_bounds__(256) void qkv_norm_rope(
    const bf16_t* __restrict__ qkv, const float* __restrict__ qgain,
    bf16_t* __restrict__ qb, bf16_t* __restrict__ kb, bf16_t* __restrict__ vt)
{
  const int lane = threadIdx.x & 31;
  const int rid  = blockIdx.x * 8 + (threadIdx.x >> 5);
  const int QROWS = BSZ * NUM_HEADS * SEQ;
  const int KROWS = BSZ * NUM_KV_HEADS * SEQ;

  if (rid < QROWS) {
    int b = rid / (NUM_HEADS*SEQ);
    int rem = rid % (NUM_HEADS*SEQ);
    int h = rem / SEQ, s = rem % SEQ;
    const bf16_t* src = qkv + (size_t)(b*SEQ + s)*NQKV + h*HEAD_DIM;
    bf16_t* dst = qb + (size_t)((b*NUM_HEADS + h)*SEQ + s)*HEAD_DIM;
    norm_rope_row(src, dst, s, qgain[h], lane);
  } else if (rid < QROWS + KROWS) {
    int r = rid - QROWS;
    int b = r / (NUM_KV_HEADS*SEQ);
    int rem = r % (NUM_KV_HEADS*SEQ);
    int kh = rem / SEQ, s = rem % SEQ;
    const bf16_t* src = qkv + (size_t)(b*SEQ + s)*NQKV + DIM + kh*HEAD_DIM;
    bf16_t* dst = kb + (size_t)((b*NUM_KV_HEADS + kh)*SEQ + s)*HEAD_DIM;
    norm_rope_row(src, dst, s, 1.0f, lane);
  } else {
    int r = rid - QROWS - KROWS;
    int b = r / (NUM_KV_HEADS*SEQ);
    int rem = r % (NUM_KV_HEADS*SEQ);
    int kh = rem / SEQ, s = rem % SEQ;
    const bf16_t* src = qkv + (size_t)(b*SEQ + s)*NQKV + DIM + KV_DIM + kh*HEAD_DIM;
    bf16_t* base = vt + (size_t)(b*NUM_KV_HEADS + kh)*HEAD_DIM*SEQ;
#pragma unroll
    for (int j = 0; j < 4; ++j) {
      int i = lane + 32*j;
      base[(size_t)i*SEQ + s] = src[i];       // V stored transposed [d][s]
    }
  }
}

// ---------------------------------------------------------------- flash attention
// grid = B*H*(SEQ/64); block = 128 (4 waves); wave owns 16 query rows, Bc = 32.
__global__ __launch_bounds__(128) void flash_attn(
    const bf16_t* __restrict__ qb, const bf16_t* __restrict__ kb,
    const bf16_t* __restrict__ vt, bf16_t* __restrict__ yb)
{
  __shared__ __align__(16) bf16_t Pl[4][16*32];
  const int tid  = threadIdx.x;
  const int lane = tid & 31, w = tid >> 5;
  const int l15  = lane & 15, hi = lane >> 4;
  const int qt   = blockIdx.x & 31;
  const int bh   = blockIdx.x >> 5;
  const int b    = bh >> 4, h = bh & 15;
  const int kvh  = h >> 2;                    // rep = 4
  const int q0   = qt*64 + w*16;

  FragB qf[4];
  {
    const bf16_t* qrow = qb + (size_t)((b*NUM_HEADS + h)*SEQ + q0 + l15)*HEAD_DIM;
#pragma unroll
    for (int dt = 0; dt < 4; ++dt) {
      qf[dt].q[0] = *(const u32x4*)&qrow[dt*32 + hi*8];
      qf[dt].q[1] = *(const u32x4*)&qrow[dt*32 + 16 + hi*8];
    }
  }
  const bf16_t* Kb = kb + (size_t)(b*NUM_KV_HEADS + kvh)*SEQ*HEAD_DIM;
  const bf16_t* Vb = vt + (size_t)(b*NUM_KV_HEADS + kvh)*HEAD_DIM*SEQ;

  const v8f vz = {0.f,0.f,0.f,0.f,0.f,0.f,0.f,0.f};
  v8f O[8];
#pragma unroll
  for (int i = 0; i < 8; ++i) O[i] = vz;
  float mr[8], lr[8];
#pragma unroll
  for (int r = 0; r < 8; ++r) { mr[r] = -1e30f; lr[r] = 0.f; }

  const float scale = 0.08838834764831845f;   // 1/sqrt(128)

  for (int t0 = 0; t0 < q0 + 16; t0 += 32) {
    // prefetch next KV tile while this one computes
    if (t0 + 32 < q0 + 16) {
      __builtin_prefetch(&Kb[(size_t)(t0 + 32 + lane)*HEAD_DIM], 0, 1);
      __builtin_prefetch(&Vb[(size_t)(lane*4)*SEQ + t0 + 32], 0, 1);
    }
    v8f s0 = vz, s1 = vz;
#pragma unroll
    for (int dt = 0; dt < 4; ++dt) {
      FragB k0f, k1f;
      const bf16_t* kr0 = &Kb[(size_t)(t0      + l15)*HEAD_DIM + dt*32 + hi*8];
      const bf16_t* kr1 = &Kb[(size_t)(t0 + 16 + l15)*HEAD_DIM + dt*32 + hi*8];
      k0f.q[0] = *(const u32x4*)kr0; k0f.q[1] = *(const u32x4*)(kr0 + 16);
      k1f.q[0] = *(const u32x4*)kr1; k1f.q[1] = *(const u32x4*)(kr1 + 16);
      s0 = __builtin_amdgcn_wmma_f32_16x16x32_bf16(false, qf[dt].v, false, k0f.v, (short)0, s0, false, false);
      s1 = __builtin_amdgcn_wmma_f32_16x16x32_bf16(false, qf[dt].v, false, k1f.v, (short)0, s1, false, false);
    }
    // online softmax per row (row M = r + 8*hi lives across 16 lanes of one half)
#pragma unroll
    for (int r = 0; r < 8; ++r) {
      const int q = q0 + r + 8*hi;
      float a0 = (t0 + l15      <= q) ? s0[r]*scale : -1e30f;
      float a1 = (t0 + 16 + l15 <= q) ? s1[r]*scale : -1e30f;
      float rm = fmaxf(a0, a1);
#pragma unroll
      for (int off = 8; off > 0; off >>= 1) rm = fmaxf(rm, __shfl_xor(rm, off, 16));
      const float mn   = fmaxf(mr[r], rm);
      const float corr = __expf(mr[r] - mn);
      const float p0   = __expf(a0 - mn);
      const float p1   = __expf(a1 - mn);
      float rs = p0 + p1;
#pragma unroll
      for (int off = 8; off > 0; off >>= 1) rs += __shfl_xor(rs, off, 16);
      lr[r] = lr[r]*corr + rs;
      mr[r] = mn;
#pragma unroll
      for (int dt = 0; dt < 8; ++dt) O[dt][r] *= corr;
      Pl[w][(r + 8*hi)*32 + l15]      = f2bf(p0);   // C-layout -> LDS (A-layout bounce)
      Pl[w][(r + 8*hi)*32 + 16 + l15] = f2bf(p1);
    }
    FragB pf;
    pf.q[0] = *(const u32x4*)&Pl[w][l15*32 + hi*8];
    pf.q[1] = *(const u32x4*)&Pl[w][l15*32 + 16 + hi*8];
#pragma unroll
    for (int dt = 0; dt < 8; ++dt) {
      FragB vf;
      const bf16_t* vr = &Vb[(size_t)(dt*16 + l15)*SEQ + t0 + hi*8];
      vf.q[0] = *(const u32x4*)vr; vf.q[1] = *(const u32x4*)(vr + 16);
      O[dt] = __builtin_amdgcn_wmma_f32_16x16x32_bf16(false, pf.v, false, vf.v, (short)0, O[dt], false, false);
    }
  }
  // normalize + store y as [b, s, h, d] bf16 rows for the output GEMM
#pragma unroll
  for (int r = 0; r < 8; ++r) {
    const int q = q0 + r + 8*hi;
    const float inv = 1.0f / lr[r];
    bf16_t* orow = yb + (size_t)(b*SEQ + q)*DIM + h*HEAD_DIM;
#pragma unroll
    for (int dt = 0; dt < 8; ++dt)
      orow[dt*16 + l15] = f2bf(O[dt][r]*inv);
  }
}

// ---------------------------------------------------------------- host
extern "C" void kernel_launch(void* const* d_in, const int* in_sizes, int n_in,
                              void* d_out, int out_size, void* d_ws, size_t ws_size,
                              hipStream_t stream) {
  const float* x  = (const float*)d_in[0];
  const float* Wq = (const float*)d_in[1];
  const float* Wk = (const float*)d_in[2];
  const float* Wv = (const float*)d_in[3];
  const float* Wo = (const float*)d_in[4];
  const float* qg = (const float*)d_in[5];
  (void)in_sizes; (void)n_in; (void)out_size; (void)ws_size;

  char* p = (char*)d_ws;
  auto take = [&](size_t bytes) { char* r = p; p += (bytes + 255) & ~(size_t)255; return r; };
  bf16_t* xb   = (bf16_t*)take((size_t)MROWS*DIM*2);
  bf16_t* Wqkv = (bf16_t*)take((size_t)NQKV*DIM*2);
  bf16_t* Wob  = (bf16_t*)take((size_t)DIM*DIM*2);
  bf16_t* qkv  = (bf16_t*)take((size_t)MROWS*NQKV*2);
  bf16_t* qbuf = (bf16_t*)take((size_t)BSZ*NUM_HEADS*SEQ*HEAD_DIM*2);
  bf16_t* kbuf = (bf16_t*)take((size_t)BSZ*NUM_KV_HEADS*SEQ*HEAD_DIM*2);
  bf16_t* vtb  = (bf16_t*)take((size_t)BSZ*NUM_KV_HEADS*HEAD_DIM*SEQ*2);
  bf16_t* yb   = (bf16_t*)take((size_t)MROWS*DIM*2);

  auto cvt = [&](const float* s, bf16_t* d, int n) {
    cvt_f32_to_bf16<<<(n + 255)/256, 256, 0, stream>>>(s, d, n);
  };
  cvt(x,  xb,   MROWS*DIM);
  cvt(Wq, Wqkv, DIM*DIM);
  cvt(Wk, Wqkv + (size_t)DIM*DIM,                      KV_DIM*DIM);
  cvt(Wv, Wqkv + (size_t)DIM*DIM + (size_t)KV_DIM*DIM, KV_DIM*DIM);
  cvt(Wo, Wob,  DIM*DIM);

  // QKV = x @ [Wq;Wk;Wv]^T  -> bf16 [8192, 3072]
  gemm_nt_bf16<false><<<dim3(NQKV/128, MROWS/128), 256, 0, stream>>>(
      xb, Wqkv, qkv, MROWS, NQKV, DIM);

  const int rows = BSZ*SEQ*(NUM_HEADS + 2*NUM_KV_HEADS);   // 196608
  qkv_norm_rope<<<rows/8, 256, 0, stream>>>(qkv, qg, qbuf, kbuf, vtb);

  flash_attn<<<BSZ*NUM_HEADS*(SEQ/64), 128, 0, stream>>>(qbuf, kbuf, vtb, yb);

  // out = y @ Wo^T -> f32
  gemm_nt_bf16<true><<<dim3(DIM/128, MROWS/128), 256, 0, stream>>>(
      yb, Wob, d_out, MROWS, DIM, DIM);
}
